// TransformerModel_9749575762421
// MI455X (gfx1250) — matrix operand order repaired and checked by reading, compiled, not run on hardware
//
#include <hip/hip_runtime.h>

typedef unsigned short u16;
typedef __attribute__((ext_vector_type(16))) __bf16       v16bf;
typedef __attribute__((ext_vector_type(8)))  float        v8f;
typedef __attribute__((ext_vector_type(4)))  unsigned int v4u;

union FragU { v4u q[2]; v16bf v; };

__device__ __forceinline__ u16 f2bf(float f) {
    unsigned int u = __float_as_uint(f);
    u += 0x7fffu + ((u >> 16) & 1u);          // round-to-nearest-even
    return (u16)(u >> 16);
}

// A-operand fragment (16x32 bf16): lanes 0-15 hold K {k..k+7, k+16..k+23},
// lanes 16-31 hold K {k+8..k+15, k+24..k+31} (ISA 7.12.2, 16-bit A 16x32).
__device__ __forceinline__ v16bf frag_ld_a(const u16* lds, int rowBase, int stride, int lane) {
    int r = rowBase + (lane & 15);
    int k = (lane >> 4) * 8;
    FragU f;
    f.q[0] = *(const v4u*)(lds + r * stride + k);
    f.q[1] = *(const v4u*)(lds + r * stride + k + 16);
    return f.v;
}

// B-operand fragment (32x16 bf16), source stored as B^T rows (i.e. W[n][k]):
// lanes 0-15 hold K {0..7, 16..23}, lanes 16-31 hold K {8..15, 24..31} (matches A).
__device__ __forceinline__ v16bf frag_ld_b(const u16* lds, int rowBase, int stride, int lane) {
    int r = rowBase + (lane & 15);
    int k = (lane >> 4) * 8;
    FragU f;
    f.q[0] = *(const v4u*)(lds + r * stride + k);
    f.q[1] = *(const v4u*)(lds + r * stride + k + 16);
    return f.v;
}

#define WMMA_BF16(a, b, c) \
    __builtin_amdgcn_wmma_f32_16x16x32_bf16(false, (a), false, (b), (short)0, (c), false, false)

// Async 16B global -> LDS copy (CDNA5 GLOBAL_LOAD_ASYNC_TO_LDS_B128, ASYNCcnt).
__device__ __forceinline__ void async_cp16(unsigned lds_byte, const void* gptr) {
    asm volatile("global_load_async_to_lds_b128 %0, %1, off"
                 :: "v"(lds_byte), "v"((unsigned long long)(size_t)gptr)
                 : "memory");
}
__device__ __forceinline__ void async_wait0() {
    asm volatile("s_wait_asynccnt 0x0" ::: "memory");
}

// ---------------------------------------------------------------------------
// GEMM: C[M,N] = A[M,K](bf16) * W[N,K]^T(bf16) [+ bias] [ReLU].
// Block tile 128x128, BK=32, 256 threads = 8 waves, wave tile 32x64
// (2x4 WMMA 16x16x32 tiles -> 8 WMMAs per K-step vs 6 fragment loads).
// Global->LDS staging uses CDNA5 async LDS copies. Flags are compile-time.
// ---------------------------------------------------------------------------
#define GSA 40   // LDS row stride in halves (padded: 80B, 16B-aligned chunks)

template <int HAS_BIAS, int RELU, int STORE_F32, int STORE_BF16>
__global__ __launch_bounds__(256) void gemm_bf16_kernel(
    const u16* __restrict__ A, const u16* __restrict__ W,
    const float* __restrict__ bias,
    float* __restrict__ Cf, u16* __restrict__ Cbf,
    int K, int ldc)
{
    __shared__ u16 As[128 * GSA];
    __shared__ u16 Bs[128 * GSA];

    const int tid  = threadIdx.x;
    const int lane = tid & 31;
    const int wave = tid >> 5;
    const long bM  = (long)blockIdx.y * 128;
    const long bN  = (long)blockIdx.x * 128;
    const int wm   = (wave & 3) * 32;     // 4 M slots of 32
    const int wn   = (wave >> 2) * 64;    // 2 N slots of 64

    const unsigned lAs = (unsigned)(size_t)(const void*)As;  // LDS byte offset
    const unsigned lBs = (unsigned)(size_t)(const void*)Bs;

    v8f acc[2][4] = {};

    // Per-thread staging chunks: 128x32 halves = 512 16B chunks per tile,
    // 2 chunks of A + 2 chunks of B per thread.
    const int c0  = tid * 2;
    const int r0  = c0 >> 2,       col0 = (c0 & 3) * 8;
    const int r1  = (c0 + 1) >> 2, col1 = ((c0 + 1) & 3) * 8;

    for (int k0 = 0; k0 < K; k0 += 32) {
        async_cp16(lAs + (r0 * GSA + col0) * 2, A + (bM + r0) * (long)K + k0 + col0);
        async_cp16(lAs + (r1 * GSA + col1) * 2, A + (bM + r1) * (long)K + k0 + col1);
        async_cp16(lBs + (r0 * GSA + col0) * 2, W + (bN + r0) * (long)K + k0 + col0);
        async_cp16(lBs + (r1 * GSA + col1) * 2, W + (bN + r1) * (long)K + k0 + col1);
        async_wait0();
        __syncthreads();

        v16bf a0 = frag_ld_a(As, wm,      GSA, lane);
        v16bf a1 = frag_ld_a(As, wm + 16, GSA, lane);
        v16bf b0 = frag_ld_b(Bs, wn,      GSA, lane);
        v16bf b1 = frag_ld_b(Bs, wn + 16, GSA, lane);
        v16bf b2 = frag_ld_b(Bs, wn + 32, GSA, lane);
        v16bf b3 = frag_ld_b(Bs, wn + 48, GSA, lane);
        acc[0][0] = WMMA_BF16(a0, b0, acc[0][0]);
        acc[0][1] = WMMA_BF16(a0, b1, acc[0][1]);
        acc[0][2] = WMMA_BF16(a0, b2, acc[0][2]);
        acc[0][3] = WMMA_BF16(a0, b3, acc[0][3]);
        acc[1][0] = WMMA_BF16(a1, b0, acc[1][0]);
        acc[1][1] = WMMA_BF16(a1, b1, acc[1][1]);
        acc[1][2] = WMMA_BF16(a1, b2, acc[1][2]);
        acc[1][3] = WMMA_BF16(a1, b3, acc[1][3]);
        __syncthreads();
    }

    // Epilogue: branch-free (flags compile-time), addresses hoisted per tile.
    const int mo    = 8 * (lane >> 4);
    const int nlane = lane & 15;
    const long mBase = bM + wm + mo;          // first of 8 rows this lane touches
    const long nBase = bN + wn + nlane;

#pragma unroll
    for (int mi = 0; mi < 2; ++mi) {
        const long rowStart = (mBase + 16 * mi) * (long)ldc + nBase;
#pragma unroll
        for (int ni = 0; ni < 4; ++ni) {
            const long tileOff = rowStart + 16 * ni;
            float biasv = 0.0f;
            if (HAS_BIAS) biasv = bias[nBase + 16 * ni];
#pragma unroll
            for (int r = 0; r < 8; ++r) {
                float v = acc[mi][ni][r];
                if (HAS_BIAS) v += biasv;
                if (RELU)     v = fmaxf(v, 0.0f);
                const long idx = tileOff + (long)r * ldc;
                if (STORE_F32)  Cf [idx] = v;
                if (STORE_BF16) Cbf[idx] = f2bf(v);
            }
        }
    }
}

// ---------------------------------------------------------------------------
// Flash attention, causal. One block = (q-tile of 64, head, batch); 128 thr.
// qkv: [S*B][3D] f32. attbf: [S*B][D] bf16 output.
// ---------------------------------------------------------------------------
#define AS_ 512
#define AB_ 8
#define AD_ 1024
#define HD_ 64
#define LDP 72   // LDS row stride in halves (144B, 16B multiple)

__global__ __launch_bounds__(128) void attn_kernel(const float* __restrict__ qkv,
                                                   u16* __restrict__ attbf)
{
    __shared__ u16 Qs [64 * LDP];
    __shared__ u16 Ks [64 * LDP];
    __shared__ u16 Vts[64 * LDP];   // V transposed: [hd][t]
    __shared__ u16 Ps [64 * LDP];

    const int tid  = threadIdx.x;
    const int lane = tid & 31;
    const int wave = tid >> 5;          // 0..3, owns rows 16*wave..16*wave+15
    const int qt   = blockIdx.x;        // 0..7
    const int h    = blockIdx.y;        // 0..15
    const int bb   = blockIdx.z;        // 0..7
    const int mo    = 8 * (lane >> 4);
    const int nlane = lane & 15;

    // Load Q tile (scaled by 1/sqrt(HD)=0.125) as bf16 into LDS.
    {
        int i = tid >> 1, c0 = (tid & 1) * 32;
        const float* qrow = qkv + ((size_t)(qt * 64 + i) * AB_ + bb) * (3 * AD_) + h * HD_;
#pragma unroll 8
        for (int j = 0; j < 32; ++j)
            Qs[i * LDP + c0 + j] = f2bf(qrow[c0 + j] * 0.125f);
    }

    v8f oacc[4] = {};
    float run_m[8], run_l[8];
#pragma unroll
    for (int r = 0; r < 8; ++r) { run_m[r] = -1e30f; run_l[r] = 0.0f; }

    for (int kt = 0; kt <= qt; ++kt) {
        // Load K tile row-major and V tile transposed.
        {
            int t = tid >> 1, c0 = (tid & 1) * 32;
            const float* krow = qkv + ((size_t)(kt * 64 + t) * AB_ + bb) * (3 * AD_) + AD_ + h * HD_;
            const float* vrow = krow + AD_;
#pragma unroll 8
            for (int j = 0; j < 32; ++j) {
                Ks[t * LDP + c0 + j]        = f2bf(krow[c0 + j]);
                Vts[(c0 + j) * LDP + t]     = f2bf(vrow[c0 + j]);
            }
        }
        __syncthreads();

        // S = Q * K^T  (16x64 per wave, 4 n-tiles, K-dim = HD = 64)
        v8f sf[4] = {};
        {
            v16bf qlo = frag_ld_a(Qs,      16 * wave, LDP, lane);
            v16bf qhi = frag_ld_a(Qs + 32, 16 * wave, LDP, lane);
#pragma unroll
            for (int nt = 0; nt < 4; ++nt) {
                v16bf klo = frag_ld_b(Ks,      16 * nt, LDP, lane);
                v16bf khi = frag_ld_b(Ks + 32, 16 * nt, LDP, lane);
                sf[nt] = WMMA_BF16(qlo, klo, sf[nt]);
                sf[nt] = WMMA_BF16(qhi, khi, sf[nt]);
            }
        }
        if (kt == qt) {
#pragma unroll
            for (int nt = 0; nt < 4; ++nt)
#pragma unroll
                for (int r = 0; r < 8; ++r) {
                    int mrow = 16 * wave + r + mo;
                    int tcol = 16 * nt + nlane;
                    if (tcol > mrow) sf[nt][r] = -1e30f;
                }
        }

        // Online softmax; each row lives in one VGPR across a 16-lane group.
#pragma unroll
        for (int r = 0; r < 8; ++r) {
            float mx = fmaxf(fmaxf(sf[0][r], sf[1][r]), fmaxf(sf[2][r], sf[3][r]));
#pragma unroll
            for (int off = 1; off < 16; off <<= 1) mx = fmaxf(mx, __shfl_xor(mx, off, 16));
            float nm = fmaxf(run_m[r], mx);
            float alpha = __expf(run_m[r] - nm);
            run_m[r] = nm;
            float rs = 0.0f;
#pragma unroll
            for (int nt = 0; nt < 4; ++nt) {
                float p = __expf(sf[nt][r] - nm);
                sf[nt][r] = p;
                rs += p;
            }
#pragma unroll
            for (int off = 1; off < 16; off <<= 1) rs += __shfl_xor(rs, off, 16);
            run_l[r] = run_l[r] * alpha + rs;
#pragma unroll
            for (int ht = 0; ht < 4; ++ht) oacc[ht][r] = oacc[ht][r] * alpha;
        }

        // Stage P (bf16) through LDS to re-shape C-layout -> A-operand layout.
#pragma unroll
        for (int nt = 0; nt < 4; ++nt)
#pragma unroll
            for (int r = 0; r < 8; ++r)
                Ps[(16 * wave + r + mo) * LDP + 16 * nt + nlane] = f2bf(sf[nt][r]);
        __syncthreads();

        // O += P * V   (A = P 16x64, B^T rows = Vts[hd][t])
        {
            v16bf plo = frag_ld_a(Ps,      16 * wave, LDP, lane);
            v16bf phi = frag_ld_a(Ps + 32, 16 * wave, LDP, lane);
#pragma unroll
            for (int ht = 0; ht < 4; ++ht) {
                v16bf vlo = frag_ld_b(Vts,      16 * ht, LDP, lane);
                v16bf vhi = frag_ld_b(Vts + 32, 16 * ht, LDP, lane);
                oacc[ht] = WMMA_BF16(plo, vlo, oacc[ht]);
                oacc[ht] = WMMA_BF16(phi, vhi, oacc[ht]);
            }
        }
        __syncthreads();
    }

    // Normalize and store bf16.
#pragma unroll
    for (int ht = 0; ht < 4; ++ht)
#pragma unroll
        for (int r = 0; r < 8; ++r) {
            int mrow = 16 * wave + r + mo;
            size_t row = (size_t)(qt * 64 + mrow) * AB_ + bb;
            int col = h * HD_ + 16 * ht + nlane;
            attbf[row * AD_ + col] = f2bf(oacc[ht][r] / run_l[r]);
        }
}

// ---------------------------------------------------------------------------
// Support kernels
// ---------------------------------------------------------------------------
__global__ __launch_bounds__(256) void embed_kernel(const int* __restrict__ src,
                                                    const float* __restrict__ emb,
                                                    u16* __restrict__ out, int D)
{
    int m = blockIdx.x;
    int tok = src[m];
    const float* row = emb + (size_t)tok * D;
#pragma unroll
    for (int i = 0; i < 4; ++i) {
        int c = threadIdx.x + i * 256;
        out[(size_t)m * D + c] = f2bf(row[c] * 32.0f);   // sqrt(1024) = 32
    }
}

__global__ __launch_bounds__(256) void pe_kernel(float* x, u16* xbf, int B, int D)
{
    int m = blockIdx.x;
    int s = m / B;
#pragma unroll
    for (int i = 0; i < 4; ++i) {
        int c = threadIdx.x + i * 256;
        float div = __expf(-(float)(c & ~1) * (9.2103403719761836f / (float)D));
        float ang = (float)s * div;
        float pe = (c & 1) ? __cosf(ang) : __sinf(ang);
        float v = x[(size_t)m * D + c] + pe;
        x[(size_t)m * D + c] = v;
        xbf[(size_t)m * D + c] = f2bf(v);
    }
}

__global__ __launch_bounds__(256) void resid_ln_kernel(
    const float* xin, const float* __restrict__ dlt,
    const float* __restrict__ g, const float* __restrict__ beta,
    float* xout, u16* __restrict__ xbf, int D)
{
    const int row = blockIdx.x;
    const int tid = threadIdx.x;
    const size_t base = (size_t)row * D;
    float t[4];
    float s = 0.f, q = 0.f;
#pragma unroll
    for (int i = 0; i < 4; ++i) {
        int c = tid * 4 + i;
        t[i] = xin[base + c] + dlt[base + c];
        s += t[i]; q += t[i] * t[i];
    }
#pragma unroll
    for (int off = 16; off >= 1; off >>= 1) {
        s += __shfl_xor(s, off, 32);
        q += __shfl_xor(q, off, 32);
    }
    __shared__ float sh[16];
    int wave = tid >> 5, lane = tid & 31;
    if (lane == 0) { sh[wave] = s; sh[8 + wave] = q; }
    __syncthreads();
    if (tid == 0) {
        float S = 0, Q = 0;
        for (int w = 0; w < 8; ++w) { S += sh[w]; Q += sh[8 + w]; }
        sh[0] = S; sh[8] = Q;
    }
    __syncthreads();
    float mu  = sh[0] / (float)D;
    float var = sh[8] / (float)D - mu * mu;
    float rs  = rsqrtf(var + 1e-5f);
#pragma unroll
    for (int i = 0; i < 4; ++i) {
        int c = tid * 4 + i;
        float y = (t[i] - mu) * rs * g[c] + beta[c];
        xout[base + c] = y;
        xbf[base + c]  = f2bf(y);
    }
}

__global__ void cast_kernel(const float* __restrict__ in, u16* __restrict__ out, long n)
{
    long i = (long)blockIdx.x * blockDim.x + threadIdx.x;
    long stride = (long)gridDim.x * blockDim.x;
    for (; i < n; i += stride) out[i] = f2bf(in[i]);
}

__global__ void tcast_kernel(const float* __restrict__ in, u16* __restrict__ out, int R, int C)
{
    long n = (long)R * C;
    long i = (long)blockIdx.x * blockDim.x + threadIdx.x;
    long stride = (long)gridDim.x * blockDim.x;
    for (; i < n; i += stride) {
        long r = i / C, c = i % C;
        out[c * (long)R + r] = f2bf(in[i]);   // out[n][k] = in[k][n]
    }
}

// ---------------------------------------------------------------------------
// Host orchestration
// ---------------------------------------------------------------------------
extern "C" void kernel_launch(void* const* d_in, const int* in_sizes, int n_in,
                              void* d_out, int out_size, void* d_ws, size_t ws_size,
                              hipStream_t stream)
{
    (void)in_sizes; (void)n_in; (void)out_size; (void)ws_size;
    const int S = 512, B = 8, D = 1024, V = 32000, L = 6, H = 16;
    const int M  = S * B;       // 4096
    const int D3 = 3 * D;       // 3072

    const int*   src   = (const int*)  d_in[0];
    const float* emb   = (const float*)d_in[1];
    const float* emean = (const float*)d_in[2];
    const float* qkvw  = (const float*)d_in[3];
    const float* qkvb  = (const float*)d_in[4];
    const float* ow    = (const float*)d_in[5];
    const float* ob    = (const float*)d_in[6];
    const float* w1    = (const float*)d_in[7];
    const float* b1    = (const float*)d_in[8];
    const float* w2    = (const float*)d_in[9];
    const float* b2    = (const float*)d_in[10];
    const float* ln1g  = (const float*)d_in[11];
    const float* ln1b  = (const float*)d_in[12];
    const float* ln2g  = (const float*)d_in[13];
    const float* ln2b  = (const float*)d_in[14];
    const float* decw  = (const float*)d_in[15];
    const float* decb  = (const float*)d_in[16];
    float* out = (float*)d_out;

    char* ws = (char*)d_ws;
    size_t off = 0;
    auto take = [&](size_t bytes) -> char* {
        char* p = ws + off;
        off += (bytes + 255) & ~(size_t)255;
        return p;
    };
    u16*   wbf    = (u16*)  take((size_t)V * D * 2);   // reused weight scratch (max: dec_w)
    float* xf     = (float*)take((size_t)M * D * 4);   // activations f32
    u16*   xbf    = (u16*)  take((size_t)M * D * 2);   // activations bf16
    float* qkvf   = (float*)take((size_t)M * D3 * 4);  // qkv projections f32
    u16*   attbf  = (u16*)  take((size_t)M * D * 2);   // attention out bf16
    float* dlt    = (float*)take((size_t)M * D * 4);   // pre-LN delta f32
    u16*   h1bf   = (u16*)  take((size_t)M * D * 2);   // ffn hidden bf16
    u16*   predbf = (u16*)  take((size_t)M * D * 2);   // pre-vocab bf16

    const dim3 gD (D  / 128, M / 128);
    const dim3 gD3(D3 / 128, M / 128);
    const dim3 gV (V  / 128, M / 128);

    // Embedding gather -> bf16
    embed_kernel<<<M, 256, 0, stream>>>(src, emb, xbf, D);
    // x = (emb*sqrt(D)) @ embed_mean.T   (no bias, f32 out)
    cast_kernel<<<2048, 256, 0, stream>>>(emean, wbf, (long)D * D);
    gemm_bf16_kernel<0, 0, 1, 0><<<gD, 256, 0, stream>>>(
        xbf, wbf, nullptr, xf, nullptr, D, D);
    // + positional encoding (also refreshes bf16 shadow)
    pe_kernel<<<M, 256, 0, stream>>>(xf, xbf, B, D);

    for (int l = 0; l < L; ++l) {
        cast_kernel<<<4096, 256, 0, stream>>>(qkvw + (size_t)l * D3 * D, wbf, (long)D3 * D);
        gemm_bf16_kernel<1, 0, 1, 0><<<gD3, 256, 0, stream>>>(
            xbf, wbf, qkvb + (size_t)l * D3, qkvf, nullptr, D, D3);

        attn_kernel<<<dim3(S / 64, H, B), 128, 0, stream>>>(qkvf, attbf);

        cast_kernel<<<4096, 256, 0, stream>>>(ow + (size_t)l * D * D, wbf, (long)D * D);
        gemm_bf16_kernel<1, 0, 1, 0><<<gD, 256, 0, stream>>>(
            attbf, wbf, ob + (size_t)l * D, dlt, nullptr, D, D);
        resid_ln_kernel<<<M, 256, 0, stream>>>(
            xf, dlt, ln1g + (size_t)l * D, ln1b + (size_t)l * D, xf, xbf, D);

        cast_kernel<<<4096, 256, 0, stream>>>(w1 + (size_t)l * D * D, wbf, (long)D * D);
        gemm_bf16_kernel<1, 1, 0, 1><<<gD, 256, 0, stream>>>(
            xbf, wbf, b1 + (size_t)l * D, nullptr, h1bf, D, D);
        cast_kernel<<<4096, 256, 0, stream>>>(w2 + (size_t)l * D * D, wbf, (long)D * D);
        gemm_bf16_kernel<1, 0, 1, 0><<<gD, 256, 0, stream>>>(
            h1bf, wbf, b2 + (size_t)l * D, dlt, nullptr, D, D);
        resid_ln_kernel<<<M, 256, 0, stream>>>(
            xf, dlt, ln2g + (size_t)l * D, ln2b + (size_t)l * D, xf, xbf, D);
    }

    // pred = x @ embed_mean  (W = embed_mean.T, so transpose-cast; bf16 out)
    tcast_kernel<<<2048, 256, 0, stream>>>(emean, wbf, D, D);
    gemm_bf16_kernel<0, 0, 0, 1><<<gD, 256, 0, stream>>>(
        xbf, wbf, nullptr, nullptr, predbf, D, D);
    // logits = pred @ dec_w.T + dec_b  -> d_out (f32, [S,B,V])
    cast_kernel<<<32768, 256, 0, stream>>>(decw, wbf, (long)V * D);
    gemm_bf16_kernel<1, 0, 1, 0><<<gV, 256, 0, stream>>>(
        predbf, wbf, decb, out, nullptr, D, V);
}
